// ConvLSTMSNN_10453950398807
// MI455X (gfx1250) — compile-verified, matrix operand order checked by reading, not run on hardware
//
#include <hip/hip_runtime.h>

// ---------------- types ----------------
typedef _Float16 half16 __attribute__((ext_vector_type(16)));
typedef float    float8 __attribute__((ext_vector_type(8)));

#define DEVI __device__ __forceinline__

DEVI float sigmoidf_(float x) { return 1.0f / (1.0f + __expf(-x)); }

// K-slot mapping for 16-bit WMMA fragments (A 16x32 / B 32x16):
// lanes 0-15 hold K = {0..7, 16..23}, lanes 16-31 hold K = {8..15, 24..31}
DEVI int kslot_of(int lane, int s) {
    return ((s < 8) ? s : s + 8) + ((lane >= 16) ? 8 : 0);
}

// ---------------- problem constants ----------------
// x: [B=64, T=100, 1, 32, 48]
constexpr int BB = 64, TT = 100;
constexpr int H1 = 32, W1 = 48, N1 = BB * H1 * W1;      // 98304
constexpr int H2 = 16, W2 = 24, N2 = BB * H2 * W2;      // 24576
constexpr int H3 = 8,  W3 = 12, N3 = BB * H3 * W3;      // 6144

// ---------------- utility kernels ----------------
__global__ void zero_f32_k(float* p, int n) {
    int i = blockIdx.x * blockDim.x + threadIdx.x;
    if (i < n) p[i] = 0.0f;
}
__global__ void zero_f16_k(_Float16* p, int n) {
    int i = blockIdx.x * blockDim.x + threadIdx.x;
    if (i < n) p[i] = (_Float16)0.0f;
}

// pack conv/fc weight [M, KREAL] (f32, row-major over (cin,ky,kx)) into WMMA
// A-fragment order: [mtile][ktile][lane(32)][slot(16)] f16, K padded to KT*32.
template <int MT, int KT, int KREAL>
__global__ void pack_w_k(const float* __restrict__ w, _Float16* __restrict__ wp) {
    int tid = blockIdx.x * blockDim.x + threadIdx.x;
    constexpr int TOTAL = MT * KT * 512;
    if (tid >= TOTAL) return;
    int mt  = tid / (KT * 512);
    int rem = tid % (KT * 512);
    int kt  = rem / 512;
    int q   = rem % 512;
    int lane = q >> 4, s = q & 15;
    int K = kt * 32 + kslot_of(lane, s);
    int M = mt * 16 + (lane & 15);
    float v = (K < KREAL) ? w[M * KREAL + K] : 0.0f;
    wp[tid] = (_Float16)v;
}

// convert x[:, t, 0, :, :] (f32) into cat1 channel 0 (f16)
__global__ void pack_x_k(const float* __restrict__ x, _Float16* __restrict__ cat1, int t) {
    int n = blockIdx.x * blockDim.x + threadIdx.x;
    if (n >= N1) return;
    int b = n / (H1 * W1);
    int r = n % (H1 * W1);
    cat1[n] = (_Float16)x[(size_t)(b * TT + t) * (H1 * W1) + r];
}

// ---------------- implicit-GEMM conv (WMMA f16 -> f32) ----------------
// z[M][N] = sum_K wp[M][K] * im2col(cat)[K][N] + bias[M]
// block = 256 thr (8 waves) arranged 4(M-groups) x 2(N-subtiles); each wave
// owns MS M-subtiles -> block tile = (64*MS) x 32. Double-buffered LDS B tile.
template <int CIN, int COUT, int HH, int WW, int KS, int KT, int MS>
__global__ __launch_bounds__(256) void conv_gemm_k(
        const _Float16* __restrict__ wp, const _Float16* __restrict__ cat,
        const float* __restrict__ bias, float* __restrict__ z) {
    constexpr int KS2   = KS * KS;
    constexpr int KREAL = CIN * KS2;
    constexpr int PAD   = KS / 2;
    constexpr int N     = BB * HH * WW;
    constexpr int HW    = HH * WW;

    __shared__ alignas(32) _Float16 lb[2][2 * 32 * 16];  // ping-pong B tiles

    const int n0     = blockIdx.x * 32;
    const int mtile0 = blockIdx.y * 4 * MS;
    const int lane   = threadIdx.x & 31;
    const int wv     = threadIdx.x >> 5;
    const int msub   = wv & 3;
    const int nsub   = wv >> 2;

    // precompute this thread's 4 gather-slot descriptors (fixed across K)
    int eIdx[4], eBase[4], eY[4], eX[4], eKs[4];
#pragma unroll
    for (int j = 0; j < 4; ++j) {
        int e     = threadIdx.x + 256 * j;
        int nsE   = e >> 9;
        int q     = e & 511;
        int laneE = q >> 4;
        int s     = q & 15;
        int n = n0 + nsE * 16 + (laneE & 15);
        int b = n / HW, rr = n % HW;
        eIdx[j]  = e;
        eKs[j]   = kslot_of(laneE, s);
        eBase[j] = b * HW;
        eY[j]    = rr / WW;
        eX[j]    = rr % WW;
    }

    auto fill = [&](int kt, int buf) {
#pragma unroll
        for (int j = 0; j < 4; ++j) {
            int K = kt * 32 + eKs[j];
            _Float16 v = (_Float16)0.0f;
            if (K < KREAL) {
                int cin = K / KS2, kr = K % KS2;
                int ky = kr / KS, kx = kr % KS;
                int yy = eY[j] + ky - PAD, xx = eX[j] + kx - PAD;
                if ((unsigned)yy < (unsigned)HH && (unsigned)xx < (unsigned)WW)
                    v = cat[(size_t)cin * N + eBase[j] + yy * WW + xx];
            }
            lb[buf][eIdx[j]] = v;
        }
    };

    const half16* pa[MS];
#pragma unroll
    for (int ms = 0; ms < MS; ++ms)
        pa[ms] = reinterpret_cast<const half16*>(wp) +
                 (size_t)(mtile0 + msub * MS + ms) * KT * 32 + lane;

    float8 acc[MS];
#pragma unroll
    for (int ms = 0; ms < MS; ++ms) acc[ms] = (float8){};

    fill(0, 0);
    for (int kt = 0; kt < KT; ++kt) {
        __syncthreads();
        half16 bf = *reinterpret_cast<const half16*>(&lb[kt & 1][(nsub * 32 + lane) * 16]);
        half16 a[MS];
#pragma unroll
        for (int ms = 0; ms < MS; ++ms) a[ms] = pa[ms][(size_t)kt * 32];
        if (kt + 1 < KT) fill(kt + 1, (kt + 1) & 1);  // overlap gather with WMMA
#pragma unroll
        for (int ms = 0; ms < MS; ++ms)
            acc[ms] = __builtin_amdgcn_wmma_f32_16x16x32_f16(false, a[ms], false, bf,
                                                             (short)0, acc[ms], false, false);
    }
    // D layout: VGPR r -> M=r (lanes 0-15) / M=r+8 (lanes 16-31), N = lane&15
    int ncol = n0 + nsub * 16 + (lane & 15);
#pragma unroll
    for (int ms = 0; ms < MS; ++ms) {
        int mbase = (mtile0 + msub * MS + ms) * 16 + ((lane >> 4) << 3);
#pragma unroll
        for (int r = 0; r < 8; ++r) {
            int M = mbase + r;
            z[(size_t)M * N + ncol] = acc[ms][r] + bias[M];
        }
    }
}

// ---------------- fused LSTM-cell + pool + spike ----------------
// one thread per pooled output (c, b, py, px): updates 4 positions' syn/mem,
// writes f16 mem back to this layer's cat buffer, pooled spike to next input.
template <int C, int HH, int WW, bool MAXPOOL, bool FCOUT>
__global__ void cell_update_k(const float* __restrict__ z,
                              float* __restrict__ syn, float* __restrict__ mem,
                              _Float16* __restrict__ mem_h,
                              _Float16* __restrict__ spk_h) {
    constexpr int N   = BB * HH * WW;
    constexpr int HW  = HH * WW;
    constexpr int PH  = HH / 2, PW = WW / 2;
    constexpr int PHW = PH * PW;
    constexpr int NP  = BB * PHW;

    int tid = blockIdx.x * blockDim.x + threadIdx.x;
    if (tid >= C * NP) return;
    int c = tid / NP, rem = tid % NP;
    int b = rem / PHW, pr = rem % PHW;
    int py = pr / PW, px = pr % PW;

    float pool = MAXPOOL ? -1e30f : 0.0f;
#pragma unroll
    for (int dy = 0; dy < 2; ++dy)
#pragma unroll
        for (int dx = 0; dx < 2; ++dx) {
            int y = py * 2 + dy, x = px * 2 + dx;
            int n = b * HW + y * WW + x;
            float zi = z[(size_t)(0 * C + c) * N + n];
            float zf = z[(size_t)(1 * C + c) * N + n];
            float zg = z[(size_t)(2 * C + c) * N + n];
            float zo = z[(size_t)(3 * C + c) * N + n];
            float s = sigmoidf_(zf) * syn[(size_t)c * N + n] +
                      sigmoidf_(zi) * tanhf(zg);
            float m = sigmoidf_(zo) * tanhf(s);
            syn[(size_t)c * N + n] = s;
            mem[(size_t)c * N + n] = m;
            mem_h[(size_t)c * N + n] = (_Float16)m;
            pool = MAXPOOL ? fmaxf(pool, m) : (pool + 0.25f * m);
        }
    float spk = (pool - 1.0f) > 0.0f ? 1.0f : 0.0f;
    if (FCOUT)
        spk_h[(size_t)b * (C * PHW) + c * PHW + pr] = (_Float16)spk;  // [B][C*PH*PW]
    else
        spk_h[(size_t)c * NP + b * PHW + pr] = (_Float16)spk;         // next cat
}

// ---------------- FC head: FC1(WMMA) + leaky + FC2 + leaky + output ----------------
__global__ __launch_bounds__(256) void fc_head_k(
        const _Float16* __restrict__ wpf, const _Float16* __restrict__ fcin,
        const float* __restrict__ fb1, const float* __restrict__ fw2,
        const float* __restrict__ fb2, float* __restrict__ mem4,
        float* __restrict__ mem5, float* __restrict__ out, int t) {
    __shared__ float spk4[BB * 128];  // 32 KB
    const int lane = threadIdx.x & 31;
    const int wv   = threadIdx.x >> 5;  // mtile 0..7 (128 features)
    const int khalf = (lane >= 16) ? 8 : 0;

    for (int nt = 0; nt < 4; ++nt) {  // 4 x 16 batch columns
        float8 acc = {};
        const half16* pa = reinterpret_cast<const half16*>(wpf) + (size_t)wv * 48 * 32 + lane;
        int bcol = nt * 16 + (lane & 15);
        for (int kt = 0; kt < 48; ++kt) {
            half16 a = pa[(size_t)kt * 32];
            half16 bf;
#pragma unroll
            for (int s = 0; s < 16; ++s) {
                int K = kt * 32 + ((s < 8) ? s : s + 8) + khalf;
                bf[s] = fcin[(size_t)bcol * 1536 + K];
            }
            acc = __builtin_amdgcn_wmma_f32_16x16x32_f16(false, a, false, bf,
                                                         (short)0, acc, false, false);
        }
        int mbase = wv * 16 + ((lane >> 4) << 3);
        int bi    = nt * 16 + (lane & 15);
#pragma unroll
        for (int r = 0; r < 8; ++r) {
            int M = mbase + r;
            float cur   = acc[r] + fb1[M];
            float prev  = mem4[bi * 128 + M];
            float reset = prev > 1.0f ? 1.0f : 0.0f;
            float mnew  = 0.9f * prev + cur - reset;  // THRESH = 1
            mem4[bi * 128 + M] = mnew;
            spk4[bi * 128 + M] = (mnew - 1.0f) > 0.0f ? 1.0f : 0.0f;
        }
    }
    __syncthreads();
    if (threadIdx.x < 128) {  // 64 batch x 2 outputs
        int bi = threadIdx.x >> 1, o = threadIdx.x & 1;
        float dot = fb2[o];
        for (int k = 0; k < 128; ++k) dot += spk4[bi * 128 + k] * fw2[o * 128 + k];
        float prev  = mem5[bi * 2 + o];
        float reset = prev > 1.0f ? 1.0f : 0.0f;
        float mnew  = 0.9f * prev + dot - reset;
        mem5[bi * 2 + o] = mnew;
        float spk = (mnew - 1.0f) > 0.0f ? 1.0f : 0.0f;
        out[t * (BB * 2) + bi * 2 + o] = spk;                     // spk_rec
        out[TT * BB * 2 + t * (BB * 2) + bi * 2 + o] = mnew;      // mem_rec
    }
}

// ---------------- host-side launcher ----------------
static inline size_t align256(size_t x) { return (x + 255) & ~(size_t)255; }

extern "C" void kernel_launch(void* const* d_in, const int* in_sizes, int n_in,
                              void* d_out, int out_size, void* d_ws, size_t ws_size,
                              hipStream_t stream) {
    const float* x   = (const float*)d_in[0];
    const float* w1  = (const float*)d_in[1];
    const float* b1  = (const float*)d_in[2];
    const float* w2  = (const float*)d_in[3];
    const float* b2  = (const float*)d_in[4];
    const float* w3  = (const float*)d_in[5];
    const float* b3  = (const float*)d_in[6];
    const float* fw1 = (const float*)d_in[7];
    const float* fb1 = (const float*)d_in[8];
    const float* fw2 = (const float*)d_in[9];
    const float* fb2 = (const float*)d_in[10];
    float* out = (float*)d_out;

    char* ws = (char*)d_ws;
    size_t off = 0;
    auto alloc = [&](size_t bytes) { char* p = ws + off; off = align256(off + bytes); return p; };

    float* syn1 = (float*)alloc((size_t)16 * N1 * 4);
    float* mem1 = (float*)alloc((size_t)16 * N1 * 4);
    float* syn2 = (float*)alloc((size_t)32 * N2 * 4);
    float* mem2 = (float*)alloc((size_t)32 * N2 * 4);
    float* syn3 = (float*)alloc((size_t)64 * N3 * 4);
    float* mem3 = (float*)alloc((size_t)64 * N3 * 4);
    float* mem4 = (float*)alloc((size_t)BB * 128 * 4);
    float* mem5 = (float*)alloc((size_t)BB * 2 * 4);
    float* z1   = (float*)alloc((size_t)64 * N1 * 4);
    float* z2   = (float*)alloc((size_t)128 * N2 * 4);
    float* z3   = (float*)alloc((size_t)256 * N3 * 4);
    _Float16* cat1 = (_Float16*)alloc((size_t)17 * N1 * 2);
    _Float16* cat2 = (_Float16*)alloc((size_t)48 * N2 * 2);
    _Float16* cat3 = (_Float16*)alloc((size_t)96 * N3 * 2);
    _Float16* fcin = (_Float16*)alloc((size_t)BB * 1536 * 2);
    _Float16* wp1  = (_Float16*)alloc((size_t)4  * 14 * 512 * 2);
    _Float16* wp2  = (_Float16*)alloc((size_t)8  * 14 * 512 * 2);
    _Float16* wp3  = (_Float16*)alloc((size_t)16 * 27 * 512 * 2);
    _Float16* wpf  = (_Float16*)alloc((size_t)8  * 48 * 512 * 2);
    (void)ws_size; (void)in_sizes; (void)n_in; (void)out_size;

    auto g = [](size_t n) { return dim3((unsigned)((n + 255) / 256)); };

    // zero all persistent state + cat buffers (mem channels must start at 0)
    zero_f32_k<<<g((size_t)16 * N1), 256, 0, stream>>>(syn1, 16 * N1);
    zero_f32_k<<<g((size_t)16 * N1), 256, 0, stream>>>(mem1, 16 * N1);
    zero_f32_k<<<g((size_t)32 * N2), 256, 0, stream>>>(syn2, 32 * N2);
    zero_f32_k<<<g((size_t)32 * N2), 256, 0, stream>>>(mem2, 32 * N2);
    zero_f32_k<<<g((size_t)64 * N3), 256, 0, stream>>>(syn3, 64 * N3);
    zero_f32_k<<<g((size_t)64 * N3), 256, 0, stream>>>(mem3, 64 * N3);
    zero_f32_k<<<g(BB * 128), 256, 0, stream>>>(mem4, BB * 128);
    zero_f32_k<<<g(BB * 2), 256, 0, stream>>>(mem5, BB * 2);
    zero_f16_k<<<g((size_t)17 * N1), 256, 0, stream>>>(cat1, 17 * N1);
    zero_f16_k<<<g((size_t)48 * N2), 256, 0, stream>>>(cat2, 48 * N2);
    zero_f16_k<<<g((size_t)96 * N3), 256, 0, stream>>>(cat3, 96 * N3);

    // pack weights into WMMA fragment order (K: L1=17*25=425, L2=48*9=432, L3=96*9=864, FC=1536)
    pack_w_k<4, 14, 425><<<g((size_t)4 * 14 * 512), 256, 0, stream>>>(w1, wp1);
    pack_w_k<8, 14, 432><<<g((size_t)8 * 14 * 512), 256, 0, stream>>>(w2, wp2);
    pack_w_k<16, 27, 864><<<g((size_t)16 * 27 * 512), 256, 0, stream>>>(w3, wp3);
    pack_w_k<8, 48, 1536><<<g((size_t)8 * 48 * 512), 256, 0, stream>>>(fw1, wpf);

    for (int t = 0; t < TT; ++t) {
        pack_x_k<<<g(N1), 256, 0, stream>>>(x, cat1, t);

        // layer 1: conv(17->64, 5x5 @32x48), MS=1 (block tile 64x32)
        conv_gemm_k<17, 64, H1, W1, 5, 14, 1>
            <<<dim3(N1 / 32, 1), 256, 0, stream>>>(wp1, cat1, b1, z1);
        cell_update_k<16, H1, W1, true, false>
            <<<g((size_t)16 * BB * (H1 / 2) * (W1 / 2)), 256, 0, stream>>>(
                z1, syn1, mem1, cat1 + (size_t)1 * N1, cat2);

        // layer 2: conv(48->128, 3x3 @16x24), MS=2 (block tile 128x32)
        conv_gemm_k<48, 128, H2, W2, 3, 14, 2>
            <<<dim3(N2 / 32, 1), 256, 0, stream>>>(wp2, cat2, b2, z2);
        cell_update_k<32, H2, W2, false, false>
            <<<g((size_t)32 * BB * (H2 / 2) * (W2 / 2)), 256, 0, stream>>>(
                z2, syn2, mem2, cat2 + (size_t)16 * N2, cat3);

        // layer 3: conv(96->256, 3x3 @8x12), MS=2 (block tile 128x32)
        conv_gemm_k<96, 256, H3, W3, 3, 27, 2>
            <<<dim3(N3 / 32, 2), 256, 0, stream>>>(wp3, cat3, b3, z3);
        cell_update_k<64, H3, W3, false, true>
            <<<g((size_t)64 * BB * (H3 / 2) * (W3 / 2)), 256, 0, stream>>>(
                z3, syn3, mem3, cat3 + (size_t)32 * N3, fcin);

        // FC head + leaky neurons + output recording
        fc_head_k<<<dim3(1), 256, 0, stream>>>(wpf, fcin, fb1, fw2, fb2,
                                               mem4, mem5, out, t);
    }
}
